// HausdorffLoss_49546742727196
// MI455X (gfx1250) — matrix operand order repaired and checked by reading, not compiled, not run on hardware
//
#include <hip/hip_runtime.h>

#define BATCH 4
#define NPTS  4096
#define DIM   128
#define TN    128   // tile rows (x) and cols (y) per block
#define LROW  20    // padded dword stride per 32-bf16 staged row (16 data dwords + 4 pad)
#define PLANE_LDS (TN * LROW)        // dwords per staged plane (2560)
#define BUF_LDS   (4 * PLANE_LDS)    // dwords per K-chunk buffer: xh,xl,yh,yl (10240)
#define PLANE_U4  (BATCH * NPTS * DIM / 8)   // uint4 elements per bf16 plane in global

typedef __attribute__((ext_vector_type(16))) __bf16       v16bf;
typedef __attribute__((ext_vector_type(8)))  float        v8f;
typedef __attribute__((ext_vector_type(8)))  unsigned int u8i;
typedef __attribute__((ext_vector_type(4)))  unsigned int u4i;
typedef __attribute__((ext_vector_type(8)))  int          i8i;
typedef __attribute__((ext_vector_type(4)))  int          i4i;

__device__ __forceinline__ unsigned int f32_to_bf16_rne(float f) {
    unsigned int u = __float_as_uint(f);
    unsigned int r = 0x7FFFu + ((u >> 16) & 1u);
    return (u + r) >> 16;
}
__device__ __forceinline__ float bf16_to_f32(unsigned int h) {
    return __uint_as_float(h << 16);
}

// ---------------------------------------------------------------------------
// TDM: issue a 2D tensor tile load (32 elems x TN rows, 2B elems, row stride
// DIM elems) from global to LDS with 64B->80B row padding (ISA 8.3/8.4).
// ---------------------------------------------------------------------------
__device__ __forceinline__ void tdm_load_tile(unsigned int lds_byte_off,
                                              const unsigned short* gtile) {
    const unsigned long long ga = (unsigned long long)(uintptr_t)gtile;
    u4i g0;
    g0[0] = 1u;                                   // count=1 (valid), no gather/restore
    g0[1] = lds_byte_off;                         // lds_addr (bytes, wg LDS base)
    g0[2] = (unsigned int)(ga & 0xFFFFFFFFu);     // global_addr[31:0]
    g0[3] = (unsigned int)((ga >> 32) & 0x01FFFFFFu) | (2u << 30);  // addr[56:32], type=2
    i8i g1;
    g1[0] = (int)((1u << 16)        // data_size = 2 bytes
                | (1u << 20)        // pad_enable
                | (3u << 22)        // pad_interval: every 16 dwords (64B)
                | (3u << 25));      // pad_amount: 4 dwords (16B)
    g1[1] = (int)((unsigned int)DIM << 16);          // tensor_dim0[15:0] in bits 63:48
    g1[2] = (int)((unsigned int)NPTS << 16);         // dim0 hi16=0; tensor_dim1[15:0]
    g1[3] = (int)(32u << 16);                        // dim1 hi16=0; tile_dim0 = 32
    g1[4] = (int)TN;                                 // tile_dim1 = 128 rows; tile_dim2 = 0
    g1[5] = (int)DIM;                                // tensor_dim0_stride[31:0]
    g1[6] = 0;                                       // stride hi; tensor_dim1_stride lo16
    g1[7] = 0;
    i4i z = {0, 0, 0, 0};
#if defined(__clang_major__) && (__clang_major__ >= 23)
    i8i z8 = {0, 0, 0, 0, 0, 0, 0, 0};
    __builtin_amdgcn_tensor_load_to_lds(g0, g1, z, z, z8, 0);
#else
    __builtin_amdgcn_tensor_load_to_lds(g0, g1, z, z, 0);
#endif
}

// ---------------------------------------------------------------------------
// Kernel 1a: one-shot f32 -> (bf16 hi, bf16 lo) plane conversion
// planes are [b][row][k], k packed 2 bf16/dword (elem 2j in low half)
// ---------------------------------------------------------------------------
__global__ void hd_convert_kernel(const float* __restrict__ x, const float* __restrict__ y,
                                  uint4* __restrict__ xh, uint4* __restrict__ xl,
                                  uint4* __restrict__ yh, uint4* __restrict__ yl) {
    const int chunks = BATCH * NPTS * DIM / 8;
    const int idx = blockIdx.x * blockDim.x + threadIdx.x;
    if (idx >= 2 * chunks) return;
    const bool isY = idx >= chunks;
    const int  c   = isY ? (idx - chunks) : idx;
    const float* src = (isY ? y : x) + (size_t)c * 8;
    float4 a = *(const float4*)(src);
    float4 b = *(const float4*)(src + 4);
    const float f[8] = {a.x, a.y, a.z, a.w, b.x, b.y, b.z, b.w};
    unsigned int hd[4], ld[4];
#pragma unroll
    for (int j = 0; j < 4; ++j) {
        unsigned int h0 = f32_to_bf16_rne(f[2 * j]);
        unsigned int h1 = f32_to_bf16_rne(f[2 * j + 1]);
        unsigned int l0 = f32_to_bf16_rne(f[2 * j]     - bf16_to_f32(h0));
        unsigned int l1 = f32_to_bf16_rne(f[2 * j + 1] - bf16_to_f32(h1));
        hd[j] = h0 | (h1 << 16);
        ld[j] = l0 | (l1 << 16);
    }
    uint4 hv = {hd[0], hd[1], hd[2], hd[3]};
    uint4 lv = {ld[0], ld[1], ld[2], ld[3]};
    if (isY) { yh[c] = hv; yl[c] = lv; }
    else     { xh[c] = hv; xl[c] = lv; }
}

// ---------------------------------------------------------------------------
// Kernel 1b: row squared-norms (exact f32) + init min buffers to +inf
// ---------------------------------------------------------------------------
__global__ void hd_prep_kernel(const float* __restrict__ x, const float* __restrict__ y,
                               float* __restrict__ x2, float* __restrict__ y2,
                               float* __restrict__ rowmin, float* __restrict__ colmin) {
    const int idx  = blockIdx.x * blockDim.x + threadIdx.x;
    const int rows = BATCH * NPTS;
    if (idx >= 2 * rows) return;
    const bool isY = idx >= rows;
    const int  r   = isY ? (idx - rows) : idx;
    const float* src = (isY ? y : x) + (size_t)r * DIM;
    float s = 0.0f;
    for (int k = 0; k < DIM; k += 4) {
        float4 v = *(const float4*)(src + k);
        s += v.x * v.x + v.y * v.y + v.z * v.z + v.w * v.w;
    }
    const float inf = __uint_as_float(0x7F800000u);
    if (isY) { y2[r] = s; colmin[r] = inf; }
    else     { x2[r] = s; rowmin[r] = inf; }
}

// ---------------------------------------------------------------------------
// Fragment gathers from LDS (wave32 layouts, ISA 7.12.2)
// ---------------------------------------------------------------------------
__device__ __forceinline__ v16bf load_fragA(const unsigned int* sm, int row, int sel) {
    const unsigned int* p = sm + row * LROW + sel * 4;
    uint4 a = *(const uint4*)(p);
    uint4 b = *(const uint4*)(p + 8);
    u8i u = {a.x, a.y, a.z, a.w, b.x, b.y, b.z, b.w};
    return __builtin_bit_cast(v16bf, u);
}
__device__ __forceinline__ v16bf load_fragB(const unsigned int* sm, int row, int sel) {
    const unsigned int* p = sm + row * LROW + sel * 4;
    uint4 a = *(const uint4*)(p);
    uint4 b = *(const uint4*)(p + 8);
    u8i u = {a.x, a.y, a.z, a.w, b.x, b.y, b.z, b.w};
    return __builtin_bit_cast(v16bf, u);
}

// ---------------------------------------------------------------------------
// Kernel 2: 128x128 D^2 tile; TDM double-buffered staging + bf16 split WMMA
// ---------------------------------------------------------------------------
__global__ __launch_bounds__(256)
void hd_tile_kernel(const unsigned short* __restrict__ xh, const unsigned short* __restrict__ xl,
                    const unsigned short* __restrict__ yh, const unsigned short* __restrict__ yl,
                    const float* __restrict__ x2, const float* __restrict__ y2,
                    unsigned int* __restrict__ rowmin, unsigned int* __restrict__ colmin) {
    // Single merged LDS block so TDM lds_addr byte offsets are compile-time known
    // (static LDS allocation starts at offset 0).
    __shared__ unsigned int smem[2 * BUF_LDS + 2 * TN];
    float* sx2 = (float*)(smem + 2 * BUF_LDS);
    float* sy2 = sx2 + TN;

    const int tn = blockIdx.x;   // tile over x rows (N)
    const int tm = blockIdx.y;   // tile over y rows (M)
    const int b  = blockIdx.z;

    const int t    = threadIdx.x;
    const int wave = t >> 5;     // 0..7
    const int lane = t & 31;
    const int lh   = lane & 15;
    const int sel  = lane >> 4;

    if (t < TN) sx2[t]      = x2[b * NPTS + tn * TN + t];
    else        sy2[t - TN] = y2[b * NPTS + tm * TN + (t - TN)];

    v8f acc[8] = {};

    const size_t xoff = (size_t)(b * NPTS + tn * TN) * DIM;  // elements
    const size_t yoff = (size_t)(b * NPTS + tm * TN) * DIM;

    // Issue K-chunk 0 into buffer 0 (wave 0; EXEC is ignored by TDM, TENSORcnt tracks).
    if (wave == 0) {
        tdm_load_tile((0 * BUF_LDS + 0 * PLANE_LDS) * 4u, xh + xoff);
        tdm_load_tile((0 * BUF_LDS + 1 * PLANE_LDS) * 4u, xl + xoff);
        tdm_load_tile((0 * BUF_LDS + 2 * PLANE_LDS) * 4u, yh + yoff);
        tdm_load_tile((0 * BUF_LDS + 3 * PLANE_LDS) * 4u, yl + yoff);
    }

#pragma unroll
    for (int i = 0; i < DIM / 32; ++i) {
        const int q = i & 1;
        __syncthreads();   // all waves finished reading buffer (i+1)&1 from iteration i-1
        if (wave == 0) {
            if (i + 1 < DIM / 32) {
                const int qn = (i + 1) & 1;
                const int kc = (i + 1) * 32;
                tdm_load_tile((qn * BUF_LDS + 0 * PLANE_LDS) * 4u, xh + xoff + kc);
                tdm_load_tile((qn * BUF_LDS + 1 * PLANE_LDS) * 4u, xl + xoff + kc);
                tdm_load_tile((qn * BUF_LDS + 2 * PLANE_LDS) * 4u, yh + yoff + kc);
                tdm_load_tile((qn * BUF_LDS + 3 * PLANE_LDS) * 4u, yl + yoff + kc);
                __builtin_amdgcn_s_wait_tensorcnt((short)4);  // in-order: chunk i landed
            } else {
                __builtin_amdgcn_s_wait_tensorcnt((short)0);
            }
        }
        __syncthreads();   // publish buffer q to all waves

        const unsigned int* bxh = smem + q * BUF_LDS + 0 * PLANE_LDS;
        const unsigned int* bxl = smem + q * BUF_LDS + 1 * PLANE_LDS;
        const unsigned int* byh = smem + q * BUF_LDS + 2 * PLANE_LDS;
        const unsigned int* byl = smem + q * BUF_LDS + 3 * PLANE_LDS;

        v16bf ah = load_fragA(bxh, wave * 16 + lh, sel);
        v16bf al = load_fragA(bxl, wave * 16 + lh, sel);
#pragma unroll
        for (int s = 0; s < 8; ++s) {
            v16bf bh = load_fragB(byh, s * 16 + lh, sel);
            v16bf bl = load_fragB(byl, s * 16 + lh, sel);
            acc[s] = __builtin_amdgcn_wmma_f32_16x16x32_bf16(false, ah, false, bh,
                                                             (short)0, acc[s], false, false);
            acc[s] = __builtin_amdgcn_wmma_f32_16x16x32_bf16(false, ah, false, bl,
                                                             (short)0, acc[s], false, false);
            acc[s] = __builtin_amdgcn_wmma_f32_16x16x32_bf16(false, al, false, bh,
                                                             (short)0, acc[s], false, false);
        }
    }

    // Epilogue: d2 = x2 + y2 - 2*xy (clamped); reductions stay in d^2 domain.
    const float inf = __uint_as_float(0x7F800000u);
    float rmin[8];
#pragma unroll
    for (int r = 0; r < 8; ++r) rmin[r] = inf;
    float cmin[8];
#pragma unroll
    for (int s = 0; s < 8; ++s) cmin[s] = inf;

#pragma unroll
    for (int s = 0; s < 8; ++s) {
        float y2v = sy2[s * 16 + lh];
#pragma unroll
        for (int r = 0; r < 8; ++r) {
            int   M   = sel ? (r + 8) : r;
            float x2v = sx2[wave * 16 + M];
            float d2  = fmaxf(x2v + y2v - 2.0f * acc[s][r], 0.0f);
            rmin[r] = fminf(rmin[r], d2);
            cmin[s] = fminf(cmin[s], d2);
        }
    }

    // Row mins: reduce across the 16 lanes of each half-wave (128 cols via 8 subtiles).
#pragma unroll
    for (int m = 1; m < 16; m <<= 1) {
#pragma unroll
        for (int r = 0; r < 8; ++r)
            rmin[r] = fminf(rmin[r], __shfl_xor(rmin[r], m, 32));
    }
    if (lh == 0) {
#pragma unroll
        for (int r = 0; r < 8; ++r) {
            int M = sel ? (r + 8) : r;
            int n = b * NPTS + tn * TN + wave * 16 + M;
            atomicMin(&rowmin[n], __float_as_uint(rmin[r]));  // nonneg f32: bit order == value order
        }
    }

    // Col mins: fold the two half-waves, then commit per column.
#pragma unroll
    for (int s = 0; s < 8; ++s)
        cmin[s] = fminf(cmin[s], __shfl_xor(cmin[s], 16, 32));
    if (sel == 0) {
#pragma unroll
        for (int s = 0; s < 8; ++s) {
            int m = b * NPTS + tm * TN + s * 16 + lh;
            atomicMin(&colmin[m], __float_as_uint(cmin[s]));
        }
    }
}

// ---------------------------------------------------------------------------
// Kernel 3: per-batch max of the directed mins, sqrt, mean over batches
// ---------------------------------------------------------------------------
__global__ void hd_final_kernel(const float* __restrict__ rowmin,
                                const float* __restrict__ colmin,
                                float* __restrict__ out) {
    __shared__ float sbuf[256];
    const int tid = threadIdx.x;
    float accum = 0.0f;
    for (int b = 0; b < BATCH; ++b) {
        float m = 0.0f;
        for (int i = tid; i < NPTS; i += 256) {
            m = fmaxf(m, rowmin[b * NPTS + i]);
            m = fmaxf(m, colmin[b * NPTS + i]);
        }
        sbuf[tid] = m;
        __syncthreads();
        for (int s = 128; s > 0; s >>= 1) {
            if (tid < s) sbuf[tid] = fmaxf(sbuf[tid], sbuf[tid + s]);
            __syncthreads();
        }
        if (tid == 0) accum += sqrtf(fmaxf(sbuf[0], 0.0f));
        __syncthreads();
    }
    if (tid == 0) *out = accum * (1.0f / BATCH);
}

// ---------------------------------------------------------------------------
extern "C" void kernel_launch(void* const* d_in, const int* in_sizes, int n_in,
                              void* d_out, int out_size, void* d_ws, size_t ws_size,
                              hipStream_t stream) {
    const float* x = (const float*)d_in[0];
    const float* y = (const float*)d_in[1];
    float* ws = (float*)d_ws;

    float* x2     = ws;                          // BATCH*NPTS floats
    float* y2     = ws + 1 * BATCH * NPTS;
    float* rowmin = ws + 2 * BATCH * NPTS;       // u32-min domain
    float* colmin = ws + 3 * BATCH * NPTS;
    uint4* planes = (uint4*)(ws + 4 * BATCH * NPTS);
    uint4* xh = planes;
    uint4* xl = xh + PLANE_U4;
    uint4* yh = xl + PLANE_U4;
    uint4* yl = yh + PLANE_U4;
    float* out = (float*)d_out;

    const int chunks = BATCH * NPTS * DIM / 8;
    hd_convert_kernel<<<(2 * chunks + 255) / 256, 256, 0, stream>>>(x, y, xh, xl, yh, yl);

    const int rows = 2 * BATCH * NPTS;
    hd_prep_kernel<<<(rows + 255) / 256, 256, 0, stream>>>(x, y, x2, y2, rowmin, colmin);

    dim3 grid(NPTS / TN, NPTS / TN, BATCH);
    hd_tile_kernel<<<grid, 256, 0, stream>>>((const unsigned short*)xh, (const unsigned short*)xl,
                                             (const unsigned short*)yh, (const unsigned short*)yl,
                                             x2, y2,
                                             (unsigned int*)rowmin, (unsigned int*)colmin);

    hd_final_kernel<<<1, 256, 0, stream>>>(rowmin, colmin, out);
}
